// _PointnetSAModuleBase_89558658056809
// MI455X (gfx1250) — matrix operand order repaired and checked by reading, not compiled, not run on hardware
//
#include <hip/hip_runtime.h>

// Problem constants (match reference)
#define B_      16
#define N_      8192
#define C_      64
#define NPOINT  1024
#define NSAMPLE 32
#define RAD2    (0.1f * 0.1f)

typedef _Float16 v8h  __attribute__((ext_vector_type(8)));
typedef _Float16 v16h __attribute__((ext_vector_type(16)));
typedef float    v8f  __attribute__((ext_vector_type(8)));

// ---------------- workspace layout (bytes) ----------------
// [0,12288)           W1 permuted+padded 64x96 f16 (cols: 0..63 feat, 64..66 xyz)
// [12288,20480)       W2 64x64 f16
// [20480,36864)       W3 128x64 f16
// [36864,2134016)     gidx int32 [B][NPOINT][NSAMPLE]
// [2134016, +16MB)    featT f16 [B][N][C]   (optional, if ws_size permits)
#define WS_W1    0
#define WS_W2    12288
#define WS_W3    20480
#define WS_GIDX  36864
#define WS_FEATT 2134016

// ================= kernel 0: prepack weights f32 -> f16 ======================
// W1 columns are permuted to match xT channel order: [feat(64) | xyz(3) | pad(29)]
__global__ __launch_bounds__(256)
void prepack_kernel(const float* __restrict__ W1, const float* __restrict__ W2,
                    const float* __restrict__ W3,
                    _Float16* __restrict__ w1p, _Float16* __restrict__ w2h,
                    _Float16* __restrict__ w3h) {
  int t = blockIdx.x * 256 + threadIdx.x;
  if (t < 64 * 96) {
    int r = t / 96, c = t % 96;
    float v;
    if (c < 64)      v = W1[r * 67 + 3 + c];        // feature channels
    else if (c < 67) v = W1[r * 67 + (c - 64)];     // xyz channels
    else             v = 0.f;                       // K padding 67 -> 96
    w1p[t] = (_Float16)v;
  } else if (t < 64 * 96 + 64 * 64) {
    int u = t - 64 * 96;
    w2h[u] = (_Float16)W2[u];
  } else if (t < 64 * 96 + 64 * 64 + 128 * 64) {
    int u = t - 64 * 96 - 64 * 64;
    w3h[u] = (_Float16)W3[u];
  }
}

// ================= kernel 0b: transpose features to (B,N,C) f16 ==============
__global__ __launch_bounds__(256)
void featT_kernel(const float* __restrict__ feat, _Float16* __restrict__ featT) {
  const int t = blockIdx.x * 256 + threadIdx.x;   // over B*N
  const int b = t / N_, n = t % N_;
  const float* F = feat + (size_t)b * C_ * N_ + n;
  _Float16 tmp[C_];
#pragma unroll
  for (int c = 0; c < C_; ++c) tmp[c] = (_Float16)F[(size_t)c * N_];
  v8h* dst = (v8h*)(featT + ((size_t)b * N_ + n) * C_);
#pragma unroll
  for (int i = 0; i < C_ / 8; ++i) {
    v8h v;
#pragma unroll
    for (int k = 0; k < 8; ++k) v[k] = tmp[i * 8 + k];
    dst[i] = v;
  }
}

// ================= kernel 1: furthest point sampling =========================
// One block (1024 thr, 32 waves) per batch. xyz + dist staged in LDS (128KB,
// fine on CDNA5's 320KB WGP LDS). Two barriers per step: wave32 shfl_xor
// argmax reduction, then one wave combines the 32 partials.
__global__ __launch_bounds__(1024)
void fps_kernel(const float* __restrict__ xyz, float* __restrict__ newxyz) {
  const int b = blockIdx.x, tid = threadIdx.x;
  const int lane = tid & 31, wv = tid >> 5;
  __shared__ float xs[N_], ys[N_], zs[N_], dist[N_];
  __shared__ float wrv[32];
  __shared__ int   wri[32];
  __shared__ float farp[3];
  const float* X = xyz + (size_t)b * N_ * 3;
  for (int k = tid; k < N_; k += 1024) {
    xs[k] = X[k * 3 + 0];
    ys[k] = X[k * 3 + 1];
    zs[k] = X[k * 3 + 2];
    dist[k] = 1e10f;
  }
  if (tid == 0) { farp[0] = X[0]; farp[1] = X[1]; farp[2] = X[2]; }
  __syncthreads();
  for (int it = 0; it < NPOINT; ++it) {
    const float fx = farp[0], fy = farp[1], fz = farp[2];
    if (tid < 3) newxyz[((size_t)b * NPOINT + it) * 3 + tid] = farp[tid];
    float bv = -1.f; int bi = 0;
#pragma unroll
    for (int jj = 0; jj < N_ / 1024; ++jj) {
      const int k = tid + jj * 1024;           // stride-1024: bank-conflict free
      float dx = xs[k] - fx, dy = ys[k] - fy, dz = zs[k] - fz;
      float d = dx * dx + dy * dy + dz * dz;
      float dm = fminf(dist[k], d);
      dist[k] = dm;
      if (dm > bv) { bv = dm; bi = k; }
    }
#pragma unroll
    for (int m = 16; m >= 1; m >>= 1) {
      float ov = __shfl_xor(bv, m, 32);
      int   oi = __shfl_xor(bi, m, 32);
      if (ov > bv) { bv = ov; bi = oi; }
    }
    if (lane == 0) { wrv[wv] = bv; wri[wv] = bi; }
    __syncthreads();
    if (wv == 0) {
      float v = wrv[lane]; int i2 = wri[lane];
#pragma unroll
      for (int m = 16; m >= 1; m >>= 1) {
        float ov = __shfl_xor(v, m, 32);
        int   oi = __shfl_xor(i2, m, 32);
        if (ov > v) { v = ov; i2 = oi; }
      }
      if (lane == 0) { farp[0] = xs[i2]; farp[1] = ys[i2]; farp[2] = zs[i2]; }
    }
    __syncthreads();
  }
}

// ================= kernel 2: ball query (one wave32 per query point) =========
__global__ __launch_bounds__(256)
void ballq_kernel(const float* __restrict__ xyz, const float* __restrict__ newxyz,
                  int* __restrict__ gidx) {
  const int wid  = threadIdx.x >> 5;
  const int lane = threadIdx.x & 31;
  const int g = blockIdx.x * 8 + wid;
  const int b = g / NPOINT, p = g % NPOINT;
  const float* X = xyz + (size_t)b * N_ * 3;
  const float qx = newxyz[((size_t)b * NPOINT + p) * 3 + 0];
  const float qy = newxyz[((size_t)b * NPOINT + p) * 3 + 1];
  const float qz = newxyz[((size_t)b * NPOINT + p) * 3 + 2];
  int* out = gidx + ((size_t)b * NPOINT + p) * NSAMPLE;
  int count = 0, first = -1;
  for (int base = 0; base < N_ && count < NSAMPLE; base += 32) {
    const int j = base + lane;
    float dx = X[j * 3 + 0] - qx;
    float dy = X[j * 3 + 1] - qy;
    float dz = X[j * 3 + 2] - qz;
    bool pred = (dx * dx + dy * dy + dz * dz) < RAD2;
    unsigned m = (unsigned)__ballot(pred);        // wave32: low 32 bits
    if (first < 0 && m) first = base + __builtin_ctz(m);
    if (pred) {
      int pos = count + __popc(m & ((1u << lane) - 1u));
      if (pos < NSAMPLE) out[pos] = j;
    }
    count += __popc(m);
  }
  if (first < 0) first = 0;
  if (lane >= count) out[lane] = first;           // pad with first index
}

// ================= WMMA fragment helpers =====================================
// A (16-bit 16x32): lane l -> row l&15, K = (l<16?0:8) + (i<8 ? i : i+8)
// B (16-bit 32x16): lane l -> col l&15, K = (l<16?0:8) + (i<8 ? i : i+8)
// D (f32 16x16):    lane l -> col l&15, rows v + (l<16?0:8), v=0..7
__device__ __forceinline__ v16h load_afrag(const _Float16* __restrict__ ap) {
  v8h lo = *(const v8h*)ap;
  v8h hi = *(const v8h*)(ap + 16);
  v16h a;
#pragma unroll
  for (int i = 0; i < 8; ++i) { a[i] = lo[i]; a[i + 8] = hi[i]; }
  return a;
}
__device__ __forceinline__ v16h load_bfrag(const _Float16* __restrict__ bp) {
  v8h lo = *(const v8h*)bp;
  v8h hi = *(const v8h*)(bp + 16);
  v16h b;
#pragma unroll
  for (int i = 0; i < 8; ++i) { b[i] = lo[i]; b[i + 8] = hi[i]; }
  return b;
}

// out = relu(W(OxKP) * inT^T + bias); activations stored transposed [col][row]
template <int O, int KP>
__device__ __forceinline__ void fc_relu(const _Float16* __restrict__ W,
                                        const float* __restrict__ bias,
                                        const _Float16* __restrict__ inT,
                                        _Float16* __restrict__ outT, int lane) {
  const int rsel = (lane < 16) ? 0 : 8;
  const int bsel = (lane < 16) ? 0 : 8;
  const int csel = lane & 15;
#pragma unroll
  for (int mt = 0; mt < O / 16; ++mt) {
#pragma unroll
    for (int nn = 0; nn < 2; ++nn) {
      v8f acc = {0.f, 0.f, 0.f, 0.f, 0.f, 0.f, 0.f, 0.f};
#pragma unroll
      for (int kt = 0; kt < KP / 32; ++kt) {
        v16h a = load_afrag(W + (size_t)(mt * 16 + csel) * KP + kt * 32 + rsel);
        v16h bm = load_bfrag(inT + (size_t)(nn * 16 + csel) * KP + kt * 32 + bsel);
        acc = __builtin_amdgcn_wmma_f32_16x16x32_f16(false, a, false, bm,
                                                     (short)0, acc, false, false);
      }
      const int r0 = mt * 16 + rsel;
      v8h o;
#pragma unroll
      for (int v = 0; v < 8; ++v) {
        float x = acc[v] + bias[r0 + v];
        o[v] = (_Float16)(x > 0.f ? x : 0.f);
      }
      *(v8h*)(outT + (size_t)(nn * 16 + csel) * 64 + r0) = o;  // next KP is 64
    }
  }
}

// ================= kernel 3: MLP (WMMA) + max-pool over samples ==============
#define WV 4  // waves (groups) per block
__global__ __launch_bounds__(128)
void mlp_kernel(const float* __restrict__ xyz, const float* __restrict__ feat,
                const _Float16* __restrict__ featT,  // may be null
                const float* __restrict__ newxyz,
                const _Float16* __restrict__ w1, const float* __restrict__ b1,
                const _Float16* __restrict__ w2, const float* __restrict__ b2,
                const _Float16* __restrict__ w3, const float* __restrict__ b3,
                const int* __restrict__ gidx, float* __restrict__ outNF) {
  __shared__ _Float16 xT[WV][32][96];   // [sample][channel]; ch 0..63 feat,
  __shared__ _Float16 h1T[WV][32][64];  //  64..66 rel-xyz, 67..95 zero pad
  __shared__ _Float16 h2T[WV][32][64];
  const int w = threadIdx.x >> 5;
  const int lane = threadIdx.x & 31;
  const int g = blockIdx.x * WV + w;
  const int b = g / NPOINT, p = g % NPOINT;

  // ---- build xT: each lane owns one sample (96 contiguous halves) ----
  {
    const int j = gidx[((size_t)b * NPOINT + p) * NSAMPLE + lane];
    const float* q = newxyz + ((size_t)b * NPOINT + p) * 3;
    const float* P = xyz + ((size_t)b * N_ + j) * 3;
    _Float16* xr = &xT[w][lane][0];
    if (featT) {
      const v8h* src = (const v8h*)(featT + ((size_t)b * N_ + j) * C_);
      v8h* drow = (v8h*)xr;
#pragma unroll
      for (int i = 0; i < C_ / 8; ++i) drow[i] = src[i];
    } else {
      const float* F = feat + (size_t)b * C_ * N_ + j;
#pragma unroll
      for (int c = 0; c < C_; ++c) xr[c] = (_Float16)F[(size_t)c * N_];
    }
    xr[64] = (_Float16)(P[0] - q[0]);
    xr[65] = (_Float16)(P[1] - q[1]);
    xr[66] = (_Float16)(P[2] - q[2]);
#pragma unroll
    for (int c = 67; c < 96; ++c) xr[c] = (_Float16)0.f;
  }
  __syncthreads();

  fc_relu<64, 96>(w1, b1, &xT[w][0][0], &h1T[w][0][0], lane);
  __syncthreads();
  fc_relu<64, 64>(w2, b2, &h1T[w][0][0], &h2T[w][0][0], lane);
  __syncthreads();

  // ---- layer 3 (128x64) fused with relu + max over 32 samples ----
  const int rsel = (lane < 16) ? 0 : 8;
  const int bsel = (lane < 16) ? 0 : 8;
  const int csel = lane & 15;
  const _Float16* inT = &h2T[w][0][0];
#pragma unroll
  for (int mt = 0; mt < 8; ++mt) {
    v8f acc0 = {0.f, 0.f, 0.f, 0.f, 0.f, 0.f, 0.f, 0.f};
    v8f acc1 = {0.f, 0.f, 0.f, 0.f, 0.f, 0.f, 0.f, 0.f};
#pragma unroll
    for (int kt = 0; kt < 2; ++kt) {
      v16h a = load_afrag(w3 + (size_t)(mt * 16 + csel) * 64 + kt * 32 + rsel);
      v16h bm0 = load_bfrag(inT + (size_t)csel * 64 + kt * 32 + bsel);
      v16h bm1 = load_bfrag(inT + (size_t)(16 + csel) * 64 + kt * 32 + bsel);
      acc0 = __builtin_amdgcn_wmma_f32_16x16x32_f16(false, a, false, bm0,
                                                    (short)0, acc0, false, false);
      acc1 = __builtin_amdgcn_wmma_f32_16x16x32_f16(false, a, false, bm1,
                                                    (short)0, acc1, false, false);
    }
    const int r0 = mt * 16 + rsel;
#pragma unroll
    for (int v = 0; v < 8; ++v) {
      const float bb = b3[r0 + v];
      float x0 = acc0[v] + bb; x0 = x0 > 0.f ? x0 : 0.f;
      float x1 = acc1[v] + bb; x1 = x1 > 0.f ? x1 : 0.f;
      float x = fmaxf(x0, x1);                 // max across the two col tiles
      x = fmaxf(x, __shfl_xor(x, 1, 32));      // max across 16 cols in tile
      x = fmaxf(x, __shfl_xor(x, 2, 32));
      x = fmaxf(x, __shfl_xor(x, 4, 32));
      x = fmaxf(x, __shfl_xor(x, 8, 32));
      if (csel == 0)
        outNF[((size_t)b * 128 + (r0 + v)) * NPOINT + p] = x;
    }
  }
}

// ================= host-side launcher ========================================
extern "C" void kernel_launch(void* const* d_in, const int* in_sizes, int n_in,
                              void* d_out, int out_size, void* d_ws, size_t ws_size,
                              hipStream_t stream) {
  (void)in_sizes; (void)n_in; (void)out_size;
  const float* xyz  = (const float*)d_in[0];
  const float* feat = (const float*)d_in[1];
  const float* W1   = (const float*)d_in[2];
  const float* b1   = (const float*)d_in[3];
  const float* W2   = (const float*)d_in[4];
  const float* b2   = (const float*)d_in[5];
  const float* W3   = (const float*)d_in[6];
  const float* b3   = (const float*)d_in[7];
  float* out = (float*)d_out;
  char* ws = (char*)d_ws;
  _Float16* w1p = (_Float16*)(ws + WS_W1);
  _Float16* w2h = (_Float16*)(ws + WS_W2);
  _Float16* w3h = (_Float16*)(ws + WS_W3);
  int* gidx = (int*)(ws + WS_GIDX);
  // transposed-f16 feature copy only if workspace is big enough (ws_size is
  // constant across calls -> deterministic path selection)
  const size_t featT_bytes = (size_t)B_ * N_ * C_ * sizeof(_Float16);
  _Float16* featT = (ws_size >= (size_t)WS_FEATT + featT_bytes)
                        ? (_Float16*)(ws + WS_FEATT) : nullptr;
  float* newxyz  = out;                            // (B, NPOINT, 3)
  float* newfeat = out + (size_t)B_ * NPOINT * 3;  // (B, 128, NPOINT)

  prepack_kernel<<<dim3((64 * 96 + 64 * 64 + 128 * 64 + 255) / 256), dim3(256), 0,
                   stream>>>(W1, W2, W3, w1p, w2h, w3h);
  if (featT)
    featT_kernel<<<dim3(B_ * N_ / 256), dim3(256), 0, stream>>>(feat, featT);
  fps_kernel<<<dim3(B_), dim3(1024), 0, stream>>>(xyz, newxyz);
  ballq_kernel<<<dim3(B_ * NPOINT / 8), dim3(256), 0, stream>>>(xyz, newxyz, gidx);
  mlp_kernel<<<dim3(B_ * NPOINT / WV), dim3(128), 0, stream>>>(
      xyz, feat, featT, newxyz, w1p, b1, w2h, b2, w3h, b3, gidx, newfeat);
}
